// Block_773094113453
// MI455X (gfx1250) — matrix operand order repaired and checked by reading, not compiled, not run on hardware
//
#include <hip/hip_runtime.h>
#include <hip/hip_bf16.h>

// Sparse 3x3x3 conv residual block (Minkowski-style) on gfx1250.
// Strategy: bf16 gather-GEMM via v_wmma_f32_16x16x32_bf16 (f32 accum),
// weights pre-packed into WMMA B-fragment layout, feats pre-converted to bf16.
// feats (100 MB) is L2-resident on MI455X (192 MB L2), so the random gathers
// are L2 traffic; bf16 halves that traffic and feeds the matrix cores.
// Each wave owns TWO 16-point M-tiles so every B fragment load feeds 2 WMMAs.
// Masking is done by redirecting the gather base pointer to a zero row
// (1 cndmask per point/k instead of 24 v_and), keeping EXEC all-ones.

#define N_PTS 262144
#define CH    96
#define KOFF  27
#define FRAGS_PER_W (KOFF * 3 * 6)          // k * cc-chunks * n-tiles
#define FRAG_ELEMS  512                      // 32 lanes * 16 bf16
#define PACK_ELEMS  (FRAGS_PER_W * FRAG_ELEMS)

typedef __attribute__((ext_vector_type(16))) __bf16 v16bf;
typedef __attribute__((ext_vector_type(4)))  __bf16 v4bf;
typedef __attribute__((ext_vector_type(8)))  float  v8f;

union FragBF {
    uint4 u[2];
    v16bf v;
};

// ---------------------------------------------------------------------------
// Pack W[k][cin][cout] (f32) into per-lane-contiguous bf16 B fragments.
// B 32x16 bf16 layout (from ISA 16-bit sparse 64x16 layout, reduced):
//   lanes 0-15 : N = lane,    elements map to K = {0..7, 16..23}
//   lanes 16-31: N = lane-16, elements map to K = {8..15, 24..31}
// packed[frag*512 + lane*16 + e], frag = (k*3 + cc)*6 + nt
// ---------------------------------------------------------------------------
__global__ __launch_bounds__(256) void pack_weights_kernel(
    const float* __restrict__ W, __bf16* __restrict__ packed)
{
    int tid = blockIdx.x * blockDim.x + threadIdx.x;
    if (tid >= PACK_ELEMS) return;
    int f    = tid >> 9;         // fragment id
    int r    = tid & 511;
    int lane = r >> 4;
    int e    = r & 15;
    int k    = f / 18;
    int rem  = f % 18;
    int cc   = rem / 6;
    int nt   = rem % 6;
    int hi   = (lane >= 16) ? 1 : 0;
    int nlo  = lane & 15;
    int cin  = cc * 32 + (e < 8 ? hi * 8 + e : 16 + hi * 8 + (e - 8));
    int cout = nt * 16 + nlo;
    packed[tid] = (__bf16)W[((size_t)k * CH + cin) * CH + cout];
}

// ---------------------------------------------------------------------------
// f32 -> bf16 conversion, 4 elements per thread (float4 load, 8B store).
// ---------------------------------------------------------------------------
__global__ __launch_bounds__(256) void cvt_bf16x4_kernel(
    const float4* __restrict__ src, v4bf* __restrict__ dst, int n4)
{
    int t = blockIdx.x * blockDim.x + threadIdx.x;
    if (t >= n4) return;
    float4 fv = src[t];
    v4bf o;
    o.x = (__bf16)fv.x; o.y = (__bf16)fv.y;
    o.z = (__bf16)fv.z; o.w = (__bf16)fv.w;
    dst[t] = o;
}

// ---------------------------------------------------------------------------
// Zero the masked-gather zero row (ws is poisoned; rewrite every call).
// ---------------------------------------------------------------------------
__global__ __launch_bounds__(128) void zero_row_kernel(__bf16* __restrict__ zr)
{
    int t = threadIdx.x;
    if (t < CH) zr[t] = (__bf16)0.0f;
}

// ---------------------------------------------------------------------------
// Sparse conv: one wave = TWO 16-point M-tiles x all 6 N-tiles (96 channels).
// A 16x32 bf16 layout (ISA table):
//   lane<16 : M = lane,    elements map to K = {0..7, 16..23}
//   lane>=16: M = lane-16, elements map to K = {8..15, 24..31}
// -> lane gathers two contiguous 8-channel (16B) segments of its point's row;
//    lane pairs (l, l+16) cover all 32 channels exactly once.
// C/D layout: lane holds column N = (lane&15)+nt*16; VGPR r holds row
//   M = r (lanes 0-15) or 8+r (lanes 16-31).
// STAGE 0: out = bf16( prelu(acc + b) )          (feeds conv2's gather)
// STAGE 1: out = f32 ( prelu(acc + b + feats) )  (final output)
// ---------------------------------------------------------------------------
template<int STAGE>
__global__ __launch_bounds__(256) void sparse_conv_wmma(
    const __bf16*        __restrict__ inBf,
    const __bf16*        __restrict__ zeroRow,
    const int*           __restrict__ idx,
    const unsigned char* __restrict__ mask,
    const __bf16*        __restrict__ packedW,
    const float*         __restrict__ bias,
    const float*         __restrict__ alphaPtr,
    const float*         __restrict__ residual,
    __bf16*              __restrict__ outBf,
    float*               __restrict__ outF32)
{
    const int lane = threadIdx.x & 31;
    const int wave = threadIdx.x >> 5;
    const int m0   = (blockIdx.x * 8 + wave) * 32;   // 2 M-tiles per wave
    const int nlo  = lane & 15;
    const int hi   = lane >> 4;          // 0 or 1
    const int p0   = m0 + nlo;           // point gathered for M-tile 0
    const int p1   = m0 + 16 + nlo;      // point gathered for M-tile 1

    const float alpha = alphaPtr[0];

    const int segA0 = hi ? 8 : 0;        // first 8-channel segment offset
    const int segA1 = hi ? 24 : 16;      // second 8-channel segment offset

    v8f zero = {};
    v8f acc0[6], acc1[6];
#pragma unroll
    for (int nt = 0; nt < 6; ++nt) { acc0[nt] = zero; acc1[nt] = zero; }

    for (int k = 0; k < KOFF; ++k) {
        // pull next k's 18 KB weight slab toward L0 while this k computes
        {
            const int kn = (k + 1 < KOFF) ? (k + 1) : k;
            __builtin_prefetch(packedW + (size_t)(kn * 18) * FRAG_ELEMS
                                        + lane * 16, 0, 3);
        }
        const int gidx0 = idx[k * N_PTS + p0];
        const int gidx1 = idx[k * N_PTS + p1];
        const bool mk0  = mask[k * N_PTS + p0] != 0;
        const bool mk1  = mask[k * N_PTS + p1] != 0;
        // masked gather = gather from a zero row (keeps EXEC all-ones,
        // 1 cndmask per point instead of per-dword ANDs)
        const __bf16* src0 = mk0 ? inBf + (size_t)gidx0 * CH : zeroRow;
        const __bf16* src1 = mk1 ? inBf + (size_t)gidx1 * CH : zeroRow;

        // issue ALL gathers for this k-step up front (one clause, in flight
        // across the B loads + 36 WMMAs below)
        FragBF a0[3], a1[3];
#pragma unroll
        for (int cc = 0; cc < 3; ++cc) {
            a0[cc].u[0] = *reinterpret_cast<const uint4*>(src0 + cc * 32 + segA0);
            a0[cc].u[1] = *reinterpret_cast<const uint4*>(src0 + cc * 32 + segA1);
            a1[cc].u[0] = *reinterpret_cast<const uint4*>(src1 + cc * 32 + segA0);
            a1[cc].u[1] = *reinterpret_cast<const uint4*>(src1 + cc * 32 + segA1);
        }

        const __bf16* wk =
            packedW + (size_t)(k * 18) * FRAG_ELEMS + lane * 16;
#pragma unroll
        for (int cc = 0; cc < 3; ++cc) {
#pragma unroll
            for (int nt = 0; nt < 6; ++nt) {
                FragBF b;
                b.u[0] = *reinterpret_cast<const uint4*>(
                    wk + (cc * 6 + nt) * FRAG_ELEMS);
                b.u[1] = *reinterpret_cast<const uint4*>(
                    wk + (cc * 6 + nt) * FRAG_ELEMS + 8);
                acc0[nt] = __builtin_amdgcn_wmma_f32_16x16x32_bf16(
                    false, a0[cc].v, false, b.v, (short)0, acc0[nt], false, false);
                acc1[nt] = __builtin_amdgcn_wmma_f32_16x16x32_bf16(
                    false, a1[cc].v, false, b.v, (short)0, acc1[nt], false, false);
            }
        }
    }

    // Epilogue: bias + PReLU (+ residual for stage 1), scatter per D layout.
#pragma unroll
    for (int mt = 0; mt < 2; ++mt) {
        const int mbase = m0 + mt * 16 + hi * 8;
#pragma unroll
        for (int nt = 0; nt < 6; ++nt) {
            const int ch = nt * 16 + nlo;
            const float bv = bias[ch];
#pragma unroll
            for (int r = 0; r < 8; ++r) {
                const int row = mbase + r;
                float v = (mt == 0 ? acc0[nt][r] : acc1[nt][r]) + bv;
                if (STAGE == 0) {
                    v = v > 0.0f ? v : alpha * v;
                    outBf[(size_t)row * CH + ch] = (__bf16)v;
                } else {
                    v += residual[(size_t)row * CH + ch];
                    v = v > 0.0f ? v : alpha * v;
                    outF32[(size_t)row * CH + ch] = v;
                }
            }
        }
    }
}

// ---------------------------------------------------------------------------
// inputs (setup_inputs order):
//  0 feats[N,CH] f32 | 1 neighbor_idx[K,N] i32 | 2 mask[K,N] bool(u8)
//  3 W1[K,CH,CH] f32 | 4 b1[CH] f32 | 5 a1[1] f32
//  6 W2[K,CH,CH] f32 | 7 b2[CH] f32 | 8 a2[1] f32
// out: [N,CH] f32
// ws layout: feats_bf16 (48MB) | x_bf16 (48MB) | packedW1 | packedW2 | zeroRow
// ---------------------------------------------------------------------------
extern "C" void kernel_launch(void* const* d_in, const int* in_sizes, int n_in,
                              void* d_out, int out_size, void* d_ws, size_t ws_size,
                              hipStream_t stream) {
    const float*         feats = (const float*)d_in[0];
    const int*           nidx  = (const int*)d_in[1];
    const unsigned char* mask  = (const unsigned char*)d_in[2];
    const float*         W1    = (const float*)d_in[3];
    const float*         b1    = (const float*)d_in[4];
    const float*         a1    = (const float*)d_in[5];
    const float*         W2    = (const float*)d_in[6];
    const float*         b2    = (const float*)d_in[7];
    const float*         a2    = (const float*)d_in[8];
    float*               out   = (float*)d_out;

    const size_t featsBytesBf = (size_t)N_PTS * CH * sizeof(__bf16);
    char* ws = (char*)d_ws;
    __bf16* featsBf = (__bf16*)ws;
    __bf16* xBf     = (__bf16*)(ws + featsBytesBf);
    __bf16* pW1     = (__bf16*)(ws + 2 * featsBytesBf);
    __bf16* pW2     = pW1 + PACK_ELEMS;
    __bf16* zeroRow = pW2 + PACK_ELEMS;

    // 1) pack both weight tensors into WMMA B fragments (bf16); zero row
    const int packBlocks = (PACK_ELEMS + 255) / 256;           // 972
    pack_weights_kernel<<<packBlocks, 256, 0, stream>>>(W1, pW1);
    pack_weights_kernel<<<packBlocks, 256, 0, stream>>>(W2, pW2);
    zero_row_kernel<<<1, 128, 0, stream>>>(zeroRow);

    // 2) feats f32 -> bf16 (gather source for conv1)
    const int n4 = N_PTS * CH / 4;                             // 6291456
    cvt_bf16x4_kernel<<<(n4 + 255) / 256, 256, 0, stream>>>(
        (const float4*)feats, (v4bf*)featsBf, n4);

    // 3) conv1: x = prelu(sum_k gather(feats)@W1k + b1, a1) -> bf16
    const int convBlocks = N_PTS / 256;                        // 1024 (8 waves/blk,
    sparse_conv_wmma<0><<<convBlocks, 256, 0, stream>>>(       //  32 points/wave)
        featsBf, zeroRow, nidx, mask, pW1, b1, a1, nullptr, xBf, nullptr);

    // 4) conv2 + residual: out = prelu(sum_k gather(x)@W2k + b2 + feats, a2)
    sparse_conv_wmma<1><<<convBlocks, 256, 0, stream>>>(
        xBf, zeroRow, nidx, mask, pW2, b2, a2, feats, nullptr, out);
}
